// MLP_87703232184483
// MI455X (gfx1250) — matrix operand order repaired and checked by reading, not compiled, not run on hardware
//
#include <hip/hip_runtime.h>

typedef __attribute__((ext_vector_type(16))) _Float16 v16h;
typedef __attribute__((ext_vector_type(8)))  _Float16 v8h;
typedef __attribute__((ext_vector_type(8)))  float    v8f;
typedef __attribute__((ext_vector_type(4)))  float    v4f;

#define S_ITEMS     10
#define IN_F        390
#define K1PAD       416    // 13 * 32
#define N1          256
#define N2          512
#define N3          128
#define ROWS        64     // batch rows per workgroup

// LDS row strides (in halves / floats), chosen so stride(bytes)/4 mod 64 == 4*odd
// -> the 16 per-lane row reads of an A-fragment land on distinct bank groups.
#define FEAT_STRIDE 424
#define H1_STRIDE   264
#define H2_STRIDE   520
#define LOGIT_STRIDE 132

// ---- fragment loaders --------------------------------------------------

// A-matrix 16x32 f16 (ISA 7.12.2): lanes 0-15: row M=lane, K {0..7,16..23};
// lanes 16-31: row M=lane-16, K {8..15,24..31}.  Two 16B loads.
__device__ inline v16h load_afrag(const _Float16* buf, int stride, int mBase,
                                  int kBase, int lane) {
    int m = mBase + (lane & 15);
    int k = kBase + ((lane & 16) ? 8 : 0);
    const _Float16* p = buf + m * stride + k;
    union { v16h v; v8h h[2]; } u;
    u.h[0] = *reinterpret_cast<const v8h*>(p);        // K +0..7
    u.h[1] = *reinterpret_cast<const v8h*>(p + 16);   // K +16..23
    return u.v;
}

// B-matrix 32x16 f16: lanes 0-15: col N=lane, K {0..7,16..23}; lanes 16-31: col
// N=lane-16, K {8..15,24..31}.  Weights stored K-major [N][Kpad] -> two 16B loads.
__device__ inline v16h load_bfrag(const _Float16* __restrict__ W, int strideK,
                                  int nBase, int kBase, int lane) {
    int n = nBase + (lane & 15);
    int k = kBase + ((lane & 16) ? 8 : 0);
    const _Float16* p = W + (long)n * strideK + k;
    union { v16h v; v8h h[2]; } u;
    u.h[0] = *reinterpret_cast<const v8h*>(p);
    u.h[1] = *reinterpret_cast<const v8h*>(p + 16);
    return u.v;
}

// ---- weight transpose + f32->f16 + K-pad:  Wt[n*Kpad+k] = W[k*N+n] -----
__global__ void transpose_pad_f16(const float* __restrict__ W,
                                  _Float16* __restrict__ Wt,
                                  int K, int N, int Kpad) {
    int idx = blockIdx.x * blockDim.x + threadIdx.x;
    if (idx >= N * Kpad) return;
    int n = idx / Kpad, k = idx - n * Kpad;
    float v = (k < K) ? W[(long)k * N + n] : 0.0f;
    Wt[(long)n * Kpad + k] = (_Float16)v;
}

// ---- fused gather -> MLP -> softmax ------------------------------------
__global__ void __launch_bounds__(256, 1)
mlp_fused_kernel(const int* __restrict__ xIdx,
                 const float* __restrict__ carE, const float* __restrict__ regE,
                 const float* __restrict__ poiE, const float* __restrict__ wkE,
                 const float* __restrict__ tmE,
                 const _Float16* __restrict__ W1t, const float* __restrict__ b1,
                 const _Float16* __restrict__ W2t, const float* __restrict__ b2,
                 const _Float16* __restrict__ W3t, const float* __restrict__ b3,
                 float* __restrict__ out) {
    __shared__ __align__(32) _Float16 sFeat[ROWS * FEAT_STRIDE]; // 54,272 B
    __shared__ __align__(32) _Float16 sH1[ROWS * H1_STRIDE];     // 33,792 B
    __shared__ __align__(32) _Float16 sH2[ROWS * H2_STRIDE];     // 66,560 B
    float* sLogit = reinterpret_cast<float*>(sFeat);             // reuse (33,792 B)

    const int tid  = threadIdx.x;
    const int lane = tid & 31;
    const int w    = tid >> 5;           // 8 waves
    const int row0 = blockIdx.x * ROWS;

    // ---- Stage 0: gather embeddings into f16 feature tile [64 x 416] ----
    for (int idx = tid; idx < ROWS * K1PAD; idx += 256) {
        int r = idx / K1PAD, c = idx - r * K1PAD;
        float v = 0.0f;
        if (c < IN_F) {
            int s = c / 39, cc = c - s * 39;
            const int* xi = xIdx + ((long)(row0 + r) * S_ITEMS + s) * 5;
            if      (cc < 16) v = carE[xi[0] * 16 + cc];
            else if (cc < 24) v = regE[xi[1] * 8  + (cc - 16)];
            else if (cc < 28) v = poiE[xi[2] * 4  + (cc - 24)];
            else if (cc < 31) v = wkE [xi[3] * 3  + (cc - 28)];
            else              v = tmE [xi[4] * 8  + (cc - 31)];
        }
        sFeat[r * FEAT_STRIDE + c] = (_Float16)v;
    }
    __syncthreads();

    const int rb    = (w & 3) * 16;              // row-block base (0..48)
    const int mBase = rb + ((lane & 16) ? 8 : 0);
    const int nSub  = lane & 15;

    // ---- Stage 1: h1 = relu(feat @ W1 + b1)   [64 x 256], K = 416 -------
    {
        const int cb0 = (w >> 2) * 8;            // 8 col tiles per wave
        v8f acc[8] = {};
        for (int ks = 0; ks < 13; ++ks) {
            v16h a = load_afrag(sFeat, FEAT_STRIDE, rb, ks * 32, lane);
#pragma unroll
            for (int t = 0; t < 8; ++t) {
                v16h bm = load_bfrag(W1t, K1PAD, (cb0 + t) * 16, ks * 32, lane);
                acc[t] = __builtin_amdgcn_wmma_f32_16x16x32_f16(
                    false, a, false, bm, (short)0, acc[t], false, false);
            }
        }
#pragma unroll
        for (int t = 0; t < 8; ++t) {
            int n = (cb0 + t) * 16 + nSub;
            float bias = b1[n];
#pragma unroll
            for (int j = 0; j < 8; ++j) {
                float v = acc[t][j] + bias;
                sH1[(mBase + j) * H1_STRIDE + n] = (_Float16)(v > 0.0f ? v : 0.0f);
            }
        }
    }
    __syncthreads();

    // ---- Stage 2: h2 = relu(h1 @ W2 + b2)   [64 x 512], K = 256 ---------
    for (int p = 0; p < 2; ++p) {
        const int cb0 = (w >> 2) * 8 + p * 16;
        v8f acc[8] = {};
        for (int ks = 0; ks < 8; ++ks) {
            v16h a = load_afrag(sH1, H1_STRIDE, rb, ks * 32, lane);
#pragma unroll
            for (int t = 0; t < 8; ++t) {
                v16h bm = load_bfrag(W2t, N1, (cb0 + t) * 16, ks * 32, lane);
                acc[t] = __builtin_amdgcn_wmma_f32_16x16x32_f16(
                    false, a, false, bm, (short)0, acc[t], false, false);
            }
        }
#pragma unroll
        for (int t = 0; t < 8; ++t) {
            int n = (cb0 + t) * 16 + nSub;
            float bias = b2[n];
#pragma unroll
            for (int j = 0; j < 8; ++j) {
                float v = acc[t][j] + bias;
                sH2[(mBase + j) * H2_STRIDE + n] = (_Float16)(v > 0.0f ? v : 0.0f);
            }
        }
    }
    __syncthreads();

    // ---- Stage 3: logits = h2 @ W3 + b3   [64 x 128], K = 512 -----------
    {
        const int cb0 = (w >> 2) * 4;            // 4 col tiles per wave
        v8f acc[4] = {};
        for (int ks = 0; ks < 16; ++ks) {
            v16h a = load_afrag(sH2, H2_STRIDE, rb, ks * 32, lane);
#pragma unroll
            for (int t = 0; t < 4; ++t) {
                v16h bm = load_bfrag(W3t, N2, (cb0 + t) * 16, ks * 32, lane);
                acc[t] = __builtin_amdgcn_wmma_f32_16x16x32_f16(
                    false, a, false, bm, (short)0, acc[t], false, false);
            }
        }
#pragma unroll
        for (int t = 0; t < 4; ++t) {
            int n = (cb0 + t) * 16 + nSub;
            float bias = b3[n];
#pragma unroll
            for (int j = 0; j < 8; ++j)
                sLogit[(mBase + j) * LOGIT_STRIDE + n] = acc[t][j] + bias;
        }
    }
    __syncthreads();

    // ---- Stage 4: softmax over 128 labels; coalesced float4 stores ------
    for (int i = 0; i < 8; ++i) {
        int r = w * 8 + i;
        v4f v = *reinterpret_cast<const v4f*>(&sLogit[r * LOGIT_STRIDE + lane * 4]);
        float m = fmaxf(fmaxf(v[0], v[1]), fmaxf(v[2], v[3]));
#pragma unroll
        for (int off = 16; off > 0; off >>= 1)
            m = fmaxf(m, __shfl_xor(m, off, 32));
        v4f e;
        float s = 0.0f;
#pragma unroll
        for (int j = 0; j < 4; ++j) { e[j] = __expf(v[j] - m); s += e[j]; }
#pragma unroll
        for (int off = 16; off > 0; off >>= 1)
            s += __shfl_xor(s, off, 32);
        float inv = 1.0f / s;
#pragma unroll
        for (int j = 0; j < 4; ++j) e[j] *= inv;
        *reinterpret_cast<v4f*>(&out[(long)(row0 + r) * N3 + lane * 4]) = e;
    }
}

// ---- host-side launch ---------------------------------------------------
extern "C" void kernel_launch(void* const* d_in, const int* in_sizes, int n_in,
                              void* d_out, int out_size, void* d_ws, size_t ws_size,
                              hipStream_t stream) {
    const int*   x    = (const int*)  d_in[0];
    const float* carE = (const float*)d_in[1];
    const float* regE = (const float*)d_in[2];
    const float* poiE = (const float*)d_in[3];
    const float* wkE  = (const float*)d_in[4];
    const float* tmE  = (const float*)d_in[5];
    const float* W1   = (const float*)d_in[6];
    const float* b1   = (const float*)d_in[7];
    const float* W2   = (const float*)d_in[8];
    const float* b2   = (const float*)d_in[9];
    const float* W3   = (const float*)d_in[10];
    const float* b3   = (const float*)d_in[11];
    float*       out  = (float*)d_out;

    // workspace: K-major padded f16 weights
    _Float16* W1t = (_Float16*)d_ws;                 // [256][416]
    _Float16* W2t = W1t + (long)N1 * K1PAD;          // [512][256]
    _Float16* W3t = W2t + (long)N2 * N1;             // [128][512]

    int t1 = N1 * K1PAD, t2 = N2 * N1, t3 = N3 * N2;
    transpose_pad_f16<<<(t1 + 255) / 256, 256, 0, stream>>>(W1, W1t, IN_F, N1, K1PAD);
    transpose_pad_f16<<<(t2 + 255) / 256, 256, 0, stream>>>(W2, W2t, N1, N2, N1);
    transpose_pad_f16<<<(t3 + 255) / 256, 256, 0, stream>>>(W3, W3t, N2, N3, N2);

    int batch = in_sizes[0] / (S_ITEMS * 5);         // 65536
    mlp_fused_kernel<<<batch / ROWS, 256, 0, stream>>>(
        x, carE, regE, poiE, wkE, tmE, W1t, b1, W2t, b2, W3t, b3, out);
}
